// SelfTMMatcher_82927228551679
// MI455X (gfx1250) — matrix operand order repaired and checked by reading, not compiled, not run on hardware
//
#include <hip/hip_runtime.h>
#include <math.h>

typedef __attribute__((ext_vector_type(16))) _Float16 v16h;
typedef __attribute__((ext_vector_type(8)))  float    v8f;

#define WTILE 64
#define NCC_BETA 1000.0f

// ---------------------------------------------------------------------------
// NCC correlation kernel: per workgroup, a 16-row x WTILE-col output strip.
// D[m,n] = sum_{c,dx} q[c,r,x0+m+dx-PADL] * t[c,n,dx]  via WMMA f16->f32,
// scattered into out rows y = r - n + PADT.
// ---------------------------------------------------------------------------
template<int KH, int KW, int LOG2KW>
__global__ __launch_bounds__(128)
void ncc_conv_kernel(const float* __restrict__ q,
                     const float* __restrict__ t,
                     float* __restrict__ S,
                     int H, int W) {
  constexpr int NKC  = (64 * KW) / 32;     // K-chunks of 32
  constexpr int W1   = WTILE + KW - 1;     // row segment incl. halo
  constexpr int PADL = (KW - 1) / 2;
  constexpr int PADT = (KH - 1) / 2;

  __shared__ alignas(32) _Float16 Bfrag[NKC * 32 * 16]; // B operand, contiguous K layout
  __shared__ _Float16 qrow[64][W1];
  __shared__ float    sqrow[W1];
  __shared__ float    outacc[16][WTILE];
  __shared__ float    boxacc[16][WTILE];
  __shared__ float    red[128];
  __shared__ float    sumT2sh;

  const int tid   = threadIdx.x;
  const int lane  = tid & 31;
  const int wave  = tid >> 5;
  const int xbase = blockIdx.x * WTILE;
  const int y0    = blockIdx.y * 16;

  // zero accumulators
  for (int i = tid; i < 16 * WTILE; i += 128) {
    outacc[i / WTILE][i % WTILE] = 0.0f;
    boxacc[i / WTILE][i % WTILE] = 0.0f;
  }

  // sum(t*t)
  {
    float p = 0.0f;
    for (int i = tid; i < 64 * KH * KW; i += 128) { float v = t[i]; p += v * v; }
    red[tid] = p;
    __syncthreads();
    for (int s = 64; s > 0; s >>= 1) {
      if (tid < s) red[tid] += red[tid + s];
      __syncthreads();
    }
    if (tid == 0) sumT2sh = red[0];
  }

  // Pre-pack template into B fragments (contiguous K layout, same as A:
  // lane l: N = l&15 ; element e -> K = 16*(l>=16) + e within the 32-chunk).
  for (int idx = tid; idx < NKC * 32; idx += 128) {
    int kc = idx >> 5;
    int l  = idx & 31;
    int n  = l & 15;                       // dy (template row), zero-pad n>=KH
    int kbase = kc * 32 + ((l & 16) ? 16 : 0);
    _Float16* dst = &Bfrag[idx * 16];
#pragma unroll
    for (int e = 0; e < 16; ++e) {
      int k  = kbase + e;
      int c  = k >> LOG2KW;
      int dx = k & (KW - 1);
      float v = (n < KH) ? t[(c * KH + n) * KW + dx] : 0.0f;
      dst[e] = (_Float16)v;
    }
  }
  __syncthreads();

  for (int r = y0 - PADT; r <= y0 + 15 + (KH - 1 - PADT); ++r) {
    __syncthreads();   // previous iteration consumers done before overwrite
    const bool rvalid = (r >= 0) && (r < H);
    if (rvalid) {
      for (int x = tid; x < W1; x += 128) {
        int g = xbase + x - PADL;
        float acc = 0.0f;
        if (g >= 0 && g < W) {
          for (int c = 0; c < 64; ++c) {
            float v = q[((size_t)c * H + r) * W + g];   // coalesced per c
            qrow[c][x] = (_Float16)v;
            acc += v * v;
          }
        } else {
          for (int c = 0; c < 64; ++c) qrow[c][x] = (_Float16)0.0f;
        }
        sqrow[x] = acc;
      }
    }
    __syncthreads();
    if (!rvalid) continue;  // block-uniform

    // box(q^2): horizontal kw-sum of this row, added to every output row it feeds
    for (int x = tid; x < WTILE; x += 128) {
      float hsum = 0.0f;
#pragma unroll
      for (int dx = 0; dx < KW; ++dx) hsum += sqrow[x + dx];
#pragma unroll
      for (int dy = 0; dy < KH; ++dy) {
        int y = r - dy + PADT;
        if (y >= y0 && y < y0 + 16) boxacc[y - y0][x] += hsum;
      }
    }

    // WMMA: wave -> one 16-wide x-tile, full K chain
    {
      const int x0   = wave * 16;
      const int m    = lane & 15;                 // A row = output x offset
      const int hsel = (lane & 16) ? 8 : 0;       // A layout half-select
      v8f acc = {};
#pragma unroll
      for (int kc = 0; kc < NKC; ++kc) {
        v16h a;
#pragma unroll
        for (int e = 0; e < 16; ++e) {
          int klocal = e + 2 * hsel;  // contiguous K = 16*(lane>=16) + e, matches B
          int k  = kc * 32 + klocal;
          int c  = k >> LOG2KW;
          int dx = k & (KW - 1);
          a[e] = qrow[c][x0 + m + dx];
        }
        v16h b = *(const v16h*)&Bfrag[(kc * 32 + lane) * 16];
        acc = __builtin_amdgcn_wmma_f32_16x16x32_f16(
            false, a, false, b, (short)0, acc, false, false);
      }
      // scatter: D element g -> (M = g + 8*(lane>=16), N = lane&15)
      int n = lane & 15;
      int y = r - n + PADT;
      if (n < KH && y >= y0 && y < y0 + 16) {
        int mrow = (lane & 16) ? 8 : 0;
#pragma unroll
        for (int g8 = 0; g8 < 8; ++g8) {
          outacc[y - y0][x0 + mrow + g8] += acc[g8];  // unique addr per lane
        }
      }
    }
  }
  __syncthreads();

  const float st2 = sumT2sh;
  for (int i = tid; i < 16 * WTILE; i += 128) {
    int y = i / WTILE, x = i % WTILE;
    float r1  = outacc[y][x];
    float box = boxacc[y][x];
    S[(size_t)(y0 + y) * W + xbase + x] = r1 / sqrtf(st2 * box);
  }
}

// ---------------------------------------------------------------------------
// Masked softmax + expectation (one workgroup per level; levels chained on
// the stream since level i's mask depends on level i-1's prediction).
// ---------------------------------------------------------------------------
__device__ inline void region4(float ay, float ax, float pad, float h, float w,
                               float& y1, float& x1, float& y2, float& x2) {
  float y = rintf(ay * h);
  float x = rintf(ax * w);
  y1 = fmaxf(0.0f, rintf(y - pad));
  x1 = fmaxf(0.0f, rintf(x - pad));
  y2 = fminf(rintf(y + 2.0f * pad), h);
  x2 = fminf(rintf(x + 2.0f * pad), w);
}

__global__ __launch_bounds__(256)
void match_kernel(const float* __restrict__ S, int H, int W,
                  int level, float fpad, float ppad,
                  float* preds, float* out) {
  __shared__ float redA[256], redB[256], redC[256];
  const int tid = threadIdx.x;

  float y1 = 0.0f, x1 = 0.0f, y2 = (float)H, x2 = (float)W;
  if (level > 0) {
    float fy1, fx1, fy2, fx2, py1, px1, py2, px2;
    region4(preds[0],         preds[3],             fpad, (float)H, (float)W, fy1, fx1, fy2, fx2);
    region4(preds[level - 1], preds[3 + level - 1], ppad, (float)H, (float)W, py1, px1, py2, px2);
    y1 = fmaxf(fy1, py1); x1 = fmaxf(fx1, px1);
    y2 = fminf(fy2, py2); x2 = fminf(fx2, px2);
  }

  const int total = H * W;
  // pass 1: max over masked s
  float m = -__builtin_inff();
  for (int i = tid; i < total; i += 256) {
    int yy = i / W, xx = i % W;
    float fi = (float)yy, fj = (float)xx;
    bool msk = (level == 0) || (fi >= y1 && fi < y2 && fj >= x1 && fj < x2);
    if (msk) m = fmaxf(m, S[i]);
  }
  redA[tid] = m;
  __syncthreads();
  for (int s = 128; s > 0; s >>= 1) {
    if (tid < s) redA[tid] = fmaxf(redA[tid], redA[tid + s]);
    __syncthreads();
  }
  const float mlogit = NCC_BETA * redA[0];
  __syncthreads();

  // pass 2: softmax sums + expected coordinates
  float sE = 0.0f, sY = 0.0f, sX = 0.0f;
  for (int i = tid; i < total; i += 256) {
    int yy = i / W, xx = i % W;
    float fi = (float)yy, fj = (float)xx;
    bool msk = (level == 0) || (fi >= y1 && fi < y2 && fj >= x1 && fj < x2);
    if (msk) {
      float e = expf(NCC_BETA * S[i] - mlogit);
      sE += e; sY += e * fi; sX += e * fj;
    }
  }
  redA[tid] = sE; redB[tid] = sY; redC[tid] = sX;
  __syncthreads();
  for (int s = 128; s > 0; s >>= 1) {
    if (tid < s) {
      redA[tid] += redA[tid + s];
      redB[tid] += redB[tid + s];
      redC[tid] += redC[tid + s];
    }
    __syncthreads();
  }
  if (tid == 0) {
    float py = redB[0] / redA[0];
    float px = redC[0] / redA[0];
    int iy = (int)rintf(py);
    int ix = (int)rintf(px);
    float conf = S[(size_t)iy * W + ix];
    float ny = py / (float)H, nx = px / (float)W;
    preds[level]     = ny;
    preds[3 + level] = nx;
    out[level * 3 + 0] = ny;
    out[level * 3 + 1] = nx;
    out[level * 3 + 2] = conf;
  }
}

// ---------------------------------------------------------------------------
extern "C" void kernel_launch(void* const* d_in, const int* in_sizes, int n_in,
                              void* d_out, int out_size, void* d_ws, size_t ws_size,
                              hipStream_t stream) {
  const float* q0 = (const float*)d_in[0];
  const float* t0 = (const float*)d_in[1];
  const float* q1 = (const float*)d_in[2];
  const float* t1 = (const float*)d_in[3];
  const float* q2 = (const float*)d_in[4];
  const float* t2 = (const float*)d_in[5];

  float* ws    = (float*)d_ws;
  float* S0    = ws;                       // 128*128
  float* S1    = S0 + 128 * 128;           // 256*256
  float* S2    = S1 + 256 * 256;           // 512*512
  float* preds = S2 + 512 * 512;           // predY[3], predX[3]
  float* out   = (float*)d_out;            // 3x3

  // independent NCC maps
  ncc_conv_kernel<4,  4, 2><<<dim3(128 / WTILE, 128 / 16), 128, 0, stream>>>(q0, t0, S0, 128, 128);
  ncc_conv_kernel<8,  8, 3><<<dim3(256 / WTILE, 256 / 16), 128, 0, stream>>>(q1, t1, S1, 256, 256);
  ncc_conv_kernel<16, 16, 4><<<dim3(512 / WTILE, 512 / 16), 128, 0, stream>>>(q2, t2, S2, 512, 512);

  // sequential matching (mask of level i uses predictions of levels 0, i-1)
  match_kernel<<<1, 256, 0, stream>>>(S0, 128, 128, 0, 0.0f, 0.0f, preds, out);
  match_kernel<<<1, 256, 0, stream>>>(S1, 256, 256, 1, 2.0f, 2.0f, preds, out);
  match_kernel<<<1, 256, 0, stream>>>(S2, 512, 512, 2, 4.0f, 2.0f, preds, out);
}
